// SageSparseLinearAttention_9088150798927
// MI455X (gfx1250) — compile-verified, matrix operand order checked by reading, not run on hardware
//
#include <hip/hip_runtime.h>
#include <hip/hip_bf16.h>

typedef __attribute__((ext_vector_type(16))) _Float16 v16h;
typedef __attribute__((ext_vector_type(8)))  float    v8f;

// ---------------- problem constants (fixed by the reference harness) --------
constexpr int B_  = 2;
constexpr int L_  = 4096;
constexpr int H_  = 16;
constexpr int D_  = 64;
constexpr int BLKQ = 128;
constexpr int BLKK = 64;
constexpr int NQ = L_ / BLKQ;   // 32
constexpr int NK = L_ / BLKK;   // 64
constexpr int TK = 16;          // top-k key blocks per query block
constexpr float SCALE = 0.125f; // 1/sqrt(64)

// ---------------- WMMA helpers ---------------------------------------------
__device__ inline v8f wmma16(v16h a, v16h b, v8f c) {
  // D = A(16x32 f16) * B(32x16 f16) + C(16x16 f32)
  return __builtin_amdgcn_wmma_f32_16x16x32_f16(
      /*neg_a=*/false, a, /*neg_b=*/false, b,
      /*c_mod=*/(short)0, c, /*reuse_a=*/false, /*reuse_b=*/false);
}

// Load a 16x32 half fragment (A-layout; B uses the mirrored layout with the
// same per-lane addressing) from LDS where the 16 "rows" (M for A, N for B)
// are runs of `stride` halves.  kc selects the 32-wide K chunk.
__device__ inline v16h load_frag(const _Float16* base, int row0, int stride, int kc) {
  const int lane = threadIdx.x & 31;
  const int grp  = lane >> 4;       // 0: K 0-7 / 16-23 ; 1: K 8-15 / 24-31
  const int r    = lane & 15;
  const _Float16* p = base + (size_t)(row0 + r) * stride + kc * 32 + grp * 8;
  v16h f;
#pragma unroll
  for (int i = 0; i < 8; ++i) f[i] = p[i];
#pragma unroll
  for (int i = 0; i < 8; ++i) f[8 + i] = p[16 + i];
  return f;
}

__device__ inline float half_max16(float v) {
#pragma unroll
  for (int m = 1; m <= 8; m <<= 1) v = fmaxf(v, __shfl_xor(v, m, 32));
  return v;
}
__device__ inline float half_sum16(float v) {
#pragma unroll
  for (int m = 1; m <= 8; m <<= 1) v += __shfl_xor(v, m, 32);
  return v;
}

// ---------------- 1) per-block mean pooling ---------------------------------
__global__ void pool_kernel(const float* __restrict__ x, float* __restrict__ xm,
                            int blkRows, int nBlk) {
  // grid: B*H*nBlk blocks, 64 threads (one per d)
  const int g = blockIdx.x;
  const int blkI = g % nBlk;
  const int h = (g / nBlk) % H_;
  const int b = g / (nBlk * H_);
  const int d = threadIdx.x;
  float s = 0.f;
  for (int r = 0; r < blkRows; ++r)
    s += x[(((size_t)b * L_ + blkI * blkRows + r) * H_ + h) * D_ + d];
  xm[(size_t)g * D_ + d] = s / (float)blkRows;
}

// ---------------- 2) pooled scores + top-k selection ------------------------
__global__ void topk_kernel(const float* __restrict__ qm, const float* __restrict__ km,
                            int* __restrict__ topk) {
  __shared__ float sc[NK];
  const int blk = blockIdx.x;          // (b*H + h)*NQ + iq
  const int bh  = blk / NQ;
  const int j   = threadIdx.x;         // 64 threads, one per key block
  const float* qp = qm + (size_t)blk * D_;
  const float* kp = km + ((size_t)bh * NK + j) * D_;
  float s = 0.f;
#pragma unroll
  for (int d = 0; d < D_; ++d) s += qp[d] * kp[d];
  sc[j] = s;                           // scale omitted: monotone for selection
  __syncthreads();
  if (j == 0) {
    for (int t = 0; t < TK; ++t) {
      float best = -3.4e38f; int bi = 0;
      for (int c = 0; c < NK; ++c)
        if (sc[c] > best) { best = sc[c]; bi = c; }
      topk[blk * TK + t] = bi;
      sc[bi] = -3.4e38f;
    }
  }
}

// ---------------- 3) block-sparse flash attention (WMMA) --------------------
__global__ __launch_bounds__(256) void sparse_attn_kernel(
    const float* __restrict__ q, const float* __restrict__ k,
    const float* __restrict__ v, const int* __restrict__ topk,
    float* __restrict__ out) {
  __shared__ __attribute__((aligned(16))) _Float16 Qs[BLKQ * D_];   // (row, d), pre-scaled
  __shared__ __attribute__((aligned(16))) _Float16 Ks[BLKK * D_];   // (key, d)
  __shared__ __attribute__((aligned(16))) _Float16 Vt[D_ * BLKK];   // (d, key)  transposed
  __shared__ __attribute__((aligned(16))) _Float16 Pb[8 * 16 * 64]; // per-wave P

  const int blk = blockIdx.x;
  const int iq = blk % NQ;
  const int h  = (blk / NQ) % H_;
  const int b  = blk / (NQ * H_);
  const int tid = threadIdx.x;
  const int wave = tid >> 5, lane = tid & 31;
  const int grp = lane >> 4, lr = lane & 15;

  { // Q tile 128x64 -> f16, scaled by 1/sqrt(D)
    const int r = tid >> 1, c0 = (tid & 1) * 32;
    const float* src = q + (((size_t)b * L_ + iq * BLKQ + r) * H_ + h) * D_ + c0;
#pragma unroll
    for (int i = 0; i < 32; ++i) Qs[r * D_ + c0 + i] = (_Float16)(src[i] * SCALE);
  }

  v8f zf = {};
  v8f sA[4], oA[4];
  float mrow[8], lrow[8];
#pragma unroll
  for (int i = 0; i < 8; ++i) { mrow[i] = -1e30f; lrow[i] = 0.f; }
#pragma unroll
  for (int nt = 0; nt < 4; ++nt) oA[nt] = zf;

  __syncthreads();
  const int m0 = wave * 16;
  _Float16* Pw = Pb + wave * 16 * 64;

  for (int t = 0; t < TK; ++t) {
    const int kb = topk[blk * TK + t];
    { // load K block row-major + V block transposed
      const int r = tid >> 2, c0 = (tid & 3) * 16;
      const float* ks = k + (((size_t)b * L_ + kb * BLKK + r) * H_ + h) * D_ + c0;
      const float* vs = v + (((size_t)b * L_ + kb * BLKK + r) * H_ + h) * D_ + c0;
#pragma unroll
      for (int i = 0; i < 16; ++i) Ks[r * D_ + c0 + i] = (_Float16)ks[i];
#pragma unroll
      for (int i = 0; i < 16; ++i) Vt[(c0 + i) * BLKK + r] = (_Float16)vs[i];
    }
    __syncthreads();

    // S = Q * K^T   (16 rows per wave, 64 keys)
#pragma unroll
    for (int nt = 0; nt < 4; ++nt) sA[nt] = zf;
#pragma unroll
    for (int kc = 0; kc < 2; ++kc) {
      v16h aq = load_frag(Qs, m0, D_, kc);
#pragma unroll
      for (int nt = 0; nt < 4; ++nt) {
        v16h bk = load_frag(Ks, nt * 16, D_, kc);
        sA[nt] = wmma16(aq, bk, sA[nt]);
      }
    }

    // online softmax on the C-layout accumulators
#pragma unroll
    for (int i = 0; i < 8; ++i) {
      float mx = -1e30f;
#pragma unroll
      for (int nt = 0; nt < 4; ++nt) mx = fmaxf(mx, sA[nt][i]);
      mx = half_max16(mx);
      const float mnew  = fmaxf(mrow[i], mx);
      const float alpha = __expf(mrow[i] - mnew);
      float psum = 0.f;
#pragma unroll
      for (int nt = 0; nt < 4; ++nt) {
        const float p = __expf(sA[nt][i] - mnew);
        psum += p;
        Pw[(i + grp * 8) * 64 + nt * 16 + lr] = (_Float16)p;
      }
      lrow[i] = lrow[i] * alpha + half_sum16(psum);
      mrow[i] = mnew;
#pragma unroll
      for (int nt = 0; nt < 4; ++nt) oA[nt][i] *= alpha;
    }

    // O += P * V
#pragma unroll
    for (int kc = 0; kc < 2; ++kc) {
      v16h ap = load_frag(Pw, 0, 64, kc);
#pragma unroll
      for (int nt = 0; nt < 4; ++nt) {
        v16h bv = load_frag(Vt, nt * 16, BLKK, kc);
        oA[nt] = wmma16(ap, bv, oA[nt]);
      }
    }
    __syncthreads();  // Ks/Vt reused next iteration
  }

  // write o_s = O / l
#pragma unroll
  for (int i = 0; i < 8; ++i) {
    const int r = m0 + i + grp * 8;
    const float inv = 1.f / lrow[i];
    float* dst = out + (((size_t)b * L_ + iq * BLKQ + r) * H_ + h) * D_;
#pragma unroll
    for (int nt = 0; nt < 4; ++nt) dst[nt * 16 + lr] = oA[nt][i] * inv;
  }
}

// ---------------- 4) linear branch: kvsum = softmax_D(K)^T V, ksum ----------
__global__ __launch_bounds__(256) void kvsum_kernel(
    const float* __restrict__ k, const float* __restrict__ v,
    float* __restrict__ kvsum, float* __restrict__ ksum) {
  constexpr int LC = 128;  // rows per chunk
  __shared__ __attribute__((aligned(16))) _Float16 ct[D_ * LC]; // (d, l) softmax(k)^T
  __shared__ __attribute__((aligned(16))) _Float16 vt[D_ * LC]; // (e, l) v^T
  const int bh = blockIdx.x;        // b*H + h
  const int h = bh % H_, b = bh / H_;
  const int tid = threadIdx.x, wave = tid >> 5;
  const int lane = tid & 31, grp = lane >> 4, lr = lane & 15;

  v8f zf = {};
  v8f acc[2]; acc[0] = zf; acc[1] = zf;
  float ks_acc = 0.f;

  for (int l0 = 0; l0 < L_; l0 += LC) {
    if (tid < LC) {
      const int l = l0 + tid;
      const float* kr = k + (((size_t)b * L_ + l) * H_ + h) * D_;
      const float* vr = v + (((size_t)b * L_ + l) * H_ + h) * D_;
      float buf[D_]; float mx = -1e30f;
#pragma unroll
      for (int d = 0; d < D_; ++d) { buf[d] = kr[d]; mx = fmaxf(mx, buf[d]); }
      float s = 0.f;
#pragma unroll
      for (int d = 0; d < D_; ++d) { buf[d] = __expf(buf[d] - mx); s += buf[d]; }
      const float inv = 1.f / s;
#pragma unroll
      for (int d = 0; d < D_; ++d) ct[d * LC + tid] = (_Float16)(buf[d] * inv);
#pragma unroll
      for (int d = 0; d < D_; ++d) vt[d * LC + tid] = (_Float16)vr[d];
    }
    __syncthreads();
    if (tid < D_) {
      for (int ll = 0; ll < LC; ++ll) ks_acc += (float)ct[tid * LC + ll];
    }
    // 64x64 output = 16 tiles of 16x16; 2 tiles per wave
#pragma unroll
    for (int ti = 0; ti < 2; ++ti) {
      const int tile = wave * 2 + ti;
      const int m0 = (tile >> 2) * 16, n0 = (tile & 3) * 16;
#pragma unroll
      for (int kc = 0; kc < LC / 32; ++kc) {
        v16h a  = load_frag(ct, m0, LC, kc);
        v16h bb = load_frag(vt, n0, LC, kc);
        acc[ti] = wmma16(a, bb, acc[ti]);
      }
    }
    __syncthreads();
  }

#pragma unroll
  for (int ti = 0; ti < 2; ++ti) {
    const int tile = wave * 2 + ti;
    const int m0 = (tile >> 2) * 16, n0 = (tile & 3) * 16;
#pragma unroll
    for (int i = 0; i < 8; ++i) {
      const int m = m0 + i + grp * 8;
      kvsum[((size_t)bh * D_ + m) * D_ + n0 + lr] = acc[ti][i];
    }
  }
  if (tid < D_) ksum[bh * D_ + tid] = ks_acc;
}

// ---------------- 5) o_l = softmax_D(q)*kvsum / denom ; += o_l W^T + b ------
__global__ __launch_bounds__(256) void final_kernel(
    const float* __restrict__ q, const float* __restrict__ W,
    const float* __restrict__ bias, const float* __restrict__ kvsum,
    const float* __restrict__ ksum, float* __restrict__ out) {
  __shared__ __attribute__((aligned(16))) _Float16 cq[BLKQ * D_]; // (row, d)
  __shared__ __attribute__((aligned(16))) _Float16 kvT[D_ * D_];  // (e, d)
  __shared__ __attribute__((aligned(16))) _Float16 Ws[D_ * D_];   // (e, d) = W row-major
  __shared__ __attribute__((aligned(16))) _Float16 ol[8 * 16 * 64];
  __shared__ float dnm[BLKQ];
  __shared__ float ksS[D_];
  __shared__ float bS[D_];

  const int blk = blockIdx.x;
  const int iq = blk % NQ;
  const int h  = (blk / NQ) % H_;
  const int b  = blk / (NQ * H_);
  const int bh = b * H_ + h;
  const int tid = threadIdx.x, wave = tid >> 5;
  const int lane = tid & 31, grp = lane >> 4, lr = lane & 15;

  if (tid < D_) { ksS[tid] = ksum[bh * D_ + tid]; bS[tid] = bias[tid]; }
#pragma unroll
  for (int i = 0; i < 16; ++i) {
    const int idx = tid * 16 + i;
    const int m = idx >> 6, n = idx & 63;
    kvT[n * D_ + m] = (_Float16)kvsum[((size_t)bh * D_ + m) * D_ + n];
    Ws[idx] = (_Float16)W[idx];  // W[e][d] already (out,in) row-major
  }
  __syncthreads();

  if (tid < BLKQ) {
    const int l = iq * BLKQ + tid;
    const float* qr = q + (((size_t)b * L_ + l) * H_ + h) * D_;
    float buf[D_]; float mx = -1e30f;
#pragma unroll
    for (int d = 0; d < D_; ++d) { buf[d] = qr[d]; mx = fmaxf(mx, buf[d]); }
    float s = 0.f;
#pragma unroll
    for (int d = 0; d < D_; ++d) { buf[d] = __expf(buf[d] - mx); s += buf[d]; }
    const float inv = 1.f / s;
    float den = 0.f;
#pragma unroll
    for (int d = 0; d < D_; ++d) {
      const float p = buf[d] * inv;
      cq[tid * D_ + d] = (_Float16)p;
      den += p * ksS[d];
    }
    dnm[tid] = den;
  }
  __syncthreads();

  v8f zf = {};
  const int m0 = wave * 16;

  // GEMM1: numer = c_q * kvsum   (B fragment from kvT: lane e, contiguous d)
  v8f o1[4];
#pragma unroll
  for (int nt = 0; nt < 4; ++nt) o1[nt] = zf;
#pragma unroll
  for (int kc = 0; kc < 2; ++kc) {
    v16h a = load_frag(cq, m0, D_, kc);
#pragma unroll
    for (int nt = 0; nt < 4; ++nt) {
      v16h bb = load_frag(kvT, nt * 16, D_, kc);
      o1[nt] = wmma16(a, bb, o1[nt]);
    }
  }

  _Float16* olw = ol + wave * 16 * 64;
#pragma unroll
  for (int i = 0; i < 8; ++i) {
    const int m = i + grp * 8;
    const float inv = 1.f / dnm[m0 + m];
#pragma unroll
    for (int nt = 0; nt < 4; ++nt)
      olw[m * 64 + nt * 16 + lr] = (_Float16)(o1[nt][i] * inv);
  }

  // GEMM2: o_proj = o_l * W^T    (B fragment from Ws: lane e, contiguous d)
  v8f o2[4];
#pragma unroll
  for (int nt = 0; nt < 4; ++nt) o2[nt] = zf;
#pragma unroll
  for (int kc = 0; kc < 2; ++kc) {
    v16h a = load_frag(olw, 0, 64, kc);
#pragma unroll
    for (int nt = 0; nt < 4; ++nt) {
      v16h bb = load_frag(Ws, nt * 16, D_, kc);
      o2[nt] = wmma16(a, bb, o2[nt]);
    }
  }

  // out += o_proj + bias
#pragma unroll
  for (int i = 0; i < 8; ++i) {
    const int m = i + grp * 8;
    const int l = iq * BLKQ + m0 + m;
    float* dst = out + (((size_t)b * L_ + l) * H_ + h) * D_;
#pragma unroll
    for (int nt = 0; nt < 4; ++nt) {
      const int e = nt * 16 + lr;
      dst[e] += o2[nt][i] + bS[e];
    }
  }
}

// ---------------- launcher ---------------------------------------------------
extern "C" void kernel_launch(void* const* d_in, const int* in_sizes, int n_in,
                              void* d_out, int out_size, void* d_ws, size_t ws_size,
                              hipStream_t stream) {
  const float* q    = (const float*)d_in[0];
  const float* k    = (const float*)d_in[1];
  const float* v    = (const float*)d_in[2];
  const float* W    = (const float*)d_in[3];
  const float* bias = (const float*)d_in[4];
  float* out = (float*)d_out;

  char* ws = (char*)d_ws;
  float* qm  = (float*)ws; ws += (size_t)B_ * H_ * NQ * D_ * sizeof(float);
  float* km  = (float*)ws; ws += (size_t)B_ * H_ * NK * D_ * sizeof(float);
  int*   tk  = (int*)ws;   ws += (size_t)B_ * H_ * NQ * TK * sizeof(int);
  float* kvs = (float*)ws; ws += (size_t)B_ * H_ * D_ * D_ * sizeof(float);
  float* ksm = (float*)ws; ws += (size_t)B_ * H_ * D_ * sizeof(float);

  pool_kernel<<<B_ * H_ * NQ, 64, 0, stream>>>(q, qm, BLKQ, NQ);
  pool_kernel<<<B_ * H_ * NK, 64, 0, stream>>>(k, km, BLKK, NK);
  topk_kernel<<<B_ * H_ * NQ, 64, 0, stream>>>(qm, km, tk);
  sparse_attn_kernel<<<B_ * H_ * NQ, 256, 0, stream>>>(q, k, v, tk, out);
  kvsum_kernel<<<B_ * H_, 256, 0, stream>>>(k, v, kvs, ksm);
  final_kernel<<<B_ * H_ * NQ, 256, 0, stream>>>(q, W, bias, kvs, ksm, out);
}
